// GraphGuidedAttention_91044716741039
// MI455X (gfx1250) — matrix operand order repaired and checked by reading, not compiled, not run on hardware
//
#include <hip/hip_runtime.h>
#include <stdint.h>

#define B_   16
#define N_   1025
#define D_   512
#define H_   8
#define HD_  64
#define NQP  1040   /* queries padded to 65*16  */
#define NKP  1056   /* keys padded to 33*32     */
#define BN_  (B_ * N_)   /* 16400 = 1025*16, divisible by 16 */

typedef __attribute__((ext_vector_type(16))) __bf16 v16bf;
typedef __attribute__((ext_vector_type(8)))  __bf16 v8bf;
typedef __attribute__((ext_vector_type(8)))  float  v8f;

// native bf16 convert (lowers to v_cvt_*bf16_f32 on gfx1250)
static __device__ __forceinline__ __bf16 tobf(float f) { return (__bf16)f; }

static __device__ __forceinline__ v8f wmma_bf16(v16bf a, v16bf b, v8f c) {
    return __builtin_amdgcn_wmma_f32_16x16x32_bf16(false, a, false, b, (short)0, c, false, false);
}

// ---------------------------------------------------------------------------
// Kernel 0: convert weights fp32 -> bf16
// ---------------------------------------------------------------------------
__global__ __launch_bounds__(256)
void k_convert(const float* __restrict__ wqkv, const float* __restrict__ wproj,
               __bf16* __restrict__ wqkvb, __bf16* __restrict__ wprojb) {
    const int nq = D_ * 3 * D_;          // 786432
    const int np = D_ * D_;              // 262144
    const int total = nq + np;
    for (int i = blockIdx.x * 256 + threadIdx.x; i < total; i += gridDim.x * 256) {
        if (i < nq) wqkvb[i] = tobf(wqkv[i]);
        else        wprojb[i - nq] = tobf(wproj[i - nq]);
    }
}

// ---------------------------------------------------------------------------
// Kernel 1: QKV projection GEMM  (x[16400,512] fp32 @ w[512,1536] bf16)
// One wave computes a 16x32 output tile: one A fragment feeds two WMMAs
// (K=512 -> 16 k-steps, 32 WMMAs/wave). B tile 32x32 staged through LDS.
// Writes: Q row-major [bh,1040,64], K row-major [bh,1056,64], V^T [bh,64,1056].
// ---------------------------------------------------------------------------
__global__ __launch_bounds__(32)
void k_qkv_gemm(const float* __restrict__ x, const __bf16* __restrict__ wb,
                __bf16* __restrict__ qb, __bf16* __restrict__ kb,
                __bf16* __restrict__ vtb) {
    __shared__ __attribute__((aligned(16))) __bf16 Bs[32 * 32];
    const int lane = threadIdx.x;
    const int n  = lane & 15;
    const int h2 = lane >> 4;
    const int n0 = blockIdx.x * 32;     // column tile in [0,1536)
    const int m0 = blockIdx.y * 16;     // row tile in [0,16400)
    v8f acc0 = {}, acc1 = {};

    for (int k0 = 0; k0 < D_; k0 += 32) {
        // stage B tile [32 x 32] into LDS (64B coalesced per lane)
        const __bf16* bp = wb + (size_t)(k0 + lane) * (3 * D_) + n0;
        v8bf r0 = *(const v8bf*)bp;
        v8bf r1 = *(const v8bf*)(bp + 8);
        v8bf r2 = *(const v8bf*)(bp + 16);
        v8bf r3 = *(const v8bf*)(bp + 24);
        __syncthreads();
        *(v8bf*)(&Bs[lane * 32])      = r0;
        *(v8bf*)(&Bs[lane * 32 + 8])  = r1;
        *(v8bf*)(&Bs[lane * 32 + 16]) = r2;
        *(v8bf*)(&Bs[lane * 32 + 24]) = r3;

        // A fragment: lane holds row m=n, K runs {8h2..8h2+7, 16+8h2..+7}
        const float* ap = x + (size_t)(m0 + n) * D_ + k0 + 8 * h2;
        float4 f0 = *(const float4*)(ap);
        float4 f1 = *(const float4*)(ap + 4);
        float4 f2 = *(const float4*)(ap + 16);
        float4 f3 = *(const float4*)(ap + 20);
        v16bf a;
        a[0]=tobf(f0.x); a[1]=tobf(f0.y); a[2]=tobf(f0.z); a[3]=tobf(f0.w);
        a[4]=tobf(f1.x); a[5]=tobf(f1.y); a[6]=tobf(f1.z); a[7]=tobf(f1.w);
        a[8]=tobf(f2.x); a[9]=tobf(f2.y); a[10]=tobf(f2.z); a[11]=tobf(f2.w);
        a[12]=tobf(f3.x); a[13]=tobf(f3.y); a[14]=tobf(f3.z); a[15]=tobf(f3.w);

        __syncthreads();
        // B fragments: lane holds cols n and n+16, K = {8h2..8h2+7, 16+8h2..+7}
        v16bf b0, b1;
        #pragma unroll
        for (int j = 0; j < 16; ++j) {
            b0[j] = Bs[((j & 8) * 2 + 8 * h2 + (j & 7)) * 32 + n];
            b1[j] = Bs[((j & 8) * 2 + 8 * h2 + (j & 7)) * 32 + 16 + n];
        }
        acc0 = wmma_bf16(a, b0, acc0);
        acc1 = wmma_bf16(a, b1, acc1);
    }

    // scatter to q/k/vt per column-tile meaning (32-col tile within one head)
    const int three = n0 >> 9;
    const int head  = (n0 >> 6) & 7;
    #pragma unroll
    for (int half = 0; half < 2; ++half) {
        const int d = (n0 & 63) + half * 16 + n;
        #pragma unroll
        for (int i = 0; i < 8; ++i) {
            int token = m0 + 8 * h2 + i;
            int bb = token / N_;
            int t  = token - bb * N_;
            int bh = bb * H_ + head;
            __bf16 val = tobf(half ? acc1[i] : acc0[i]);
            if (three == 0)      qb[((size_t)bh * NQP + t) * HD_ + d] = val;
            else if (three == 1) kb[((size_t)bh * NKP + t) * HD_ + d] = val;
            else                 vtb[((size_t)bh * HD_ + d) * NKP + t] = val;
        }
    }
}

// ---------------------------------------------------------------------------
// Kernel 2: masked flash attention, transposed-S formulation.
//   S^T (16 keys x 16 queries) = K_tile(16x64) @ Q^T(64x16)   -> 2 WMMA
//   row softmax = 8 in-lane + one shfl_xor(16)
//   O^T (64 dims x 16 queries) += V^T_slice(16x32) @ P^T(32x16) -> 4 WMMA / 32 keys
// One wave handles one 16-query tile for one (b,h).
// ---------------------------------------------------------------------------
__global__ __launch_bounds__(32)
void k_attn(const __bf16* __restrict__ qb, const __bf16* __restrict__ kb,
            const __bf16* __restrict__ vtb, const float* __restrict__ mask,
            __bf16* __restrict__ ob) {
    const int lane = threadIdx.x;
    const int n  = lane & 15;
    const int h2 = lane >> 4;
    const int bh = blockIdx.y;            // 0..127
    const int bb = bh >> 3;
    const int head = bh & 7;
    const int q0 = blockIdx.x * 16;
    const int query = q0 + n;

    // Q^T B-fragments, constant across key loop: lane reads its query's row
    const __bf16* qrow = qb + ((size_t)bh * NQP + query) * HD_;
    v16bf bq0, bq1;
    {
        v8bf r0 = *(const v8bf*)(qrow + 8 * h2);
        v8bf r1 = *(const v8bf*)(qrow + 16 + 8 * h2);
        v8bf r2 = *(const v8bf*)(qrow + 32 + 8 * h2);
        v8bf r3 = *(const v8bf*)(qrow + 48 + 8 * h2);
        #pragma unroll
        for (int j = 0; j < 8; ++j) {
            bq0[j] = r0[j]; bq0[8 + j] = r1[j];
            bq1[j] = r2[j]; bq1[8 + j] = r3[j];
        }
    }

    const __bf16* kbase = kb  + (size_t)bh * NKP * HD_;
    const __bf16* vbase = vtb + (size_t)bh * HD_ * NKP;
    const bool qok = query < N_;
    const float* mrow = mask + (size_t)(qok ? query : N_ - 1) * N_;

    float mrun = -1e30f, lrun = 0.f;
    v8f ot[4] = {};

    for (int kt = 0; kt < NKP; kt += 32) {
        const __bf16* krow0 = kbase + (size_t)(kt + n) * HD_;
        const __bf16* krow1 = kbase + (size_t)(kt + 16 + n) * HD_;
        __builtin_prefetch((const void*)(krow0 + 32 * HD_), 0, 1);

        // K A-fragments: subtile 0 = keys kt..kt+15, subtile 1 = kt+16..kt+31
        v16bf ka00, ka01, ka10, ka11;
        {
            v8bf a0 = *(const v8bf*)(krow0 + 8 * h2);
            v8bf a1 = *(const v8bf*)(krow0 + 16 + 8 * h2);
            v8bf a2 = *(const v8bf*)(krow0 + 32 + 8 * h2);
            v8bf a3 = *(const v8bf*)(krow0 + 48 + 8 * h2);
            v8bf b0 = *(const v8bf*)(krow1 + 8 * h2);
            v8bf b1 = *(const v8bf*)(krow1 + 16 + 8 * h2);
            v8bf b2 = *(const v8bf*)(krow1 + 32 + 8 * h2);
            v8bf b3 = *(const v8bf*)(krow1 + 48 + 8 * h2);
            #pragma unroll
            for (int j = 0; j < 8; ++j) {
                ka00[j] = a0[j]; ka00[8 + j] = a1[j];
                ka01[j] = a2[j]; ka01[8 + j] = a3[j];
                ka10[j] = b0[j]; ka10[8 + j] = b1[j];
                ka11[j] = b2[j]; ka11[8 + j] = b3[j];
            }
        }

        v8f st0 = {}, st1 = {};
        st0 = wmma_bf16(ka00, bq0, st0);
        st0 = wmma_bf16(ka01, bq1, st0);
        st1 = wmma_bf16(ka10, bq0, st1);
        st1 = wmma_bf16(ka11, bq1, st1);

        // scale + mask (lane holds keys kt+8h2+i and kt+16+8h2+i of its query)
        float s0[8], s1[8];
        #pragma unroll
        for (int i = 0; i < 8; ++i) {
            int key0 = kt + 8 * h2 + i;
            int key1 = key0 + 16;
            float v0 = st0[i] * 0.125f;
            float v1 = st1[i] * 0.125f;
            s0[i] = (qok && key0 < N_ && mrow[key0] != 0.f) ? v0 : -1e30f;
            s1[i] = (qok && key1 < N_ && mrow[key1] != 0.f) ? v1 : -1e30f;
        }

        // online softmax: in-lane max over 16, half-wave exchange over lanes ^16
        float mloc = -1e30f;
        #pragma unroll
        for (int i = 0; i < 8; ++i) mloc = fmaxf(mloc, fmaxf(s0[i], s1[i]));
        mloc = fmaxf(mloc, __shfl_xor(mloc, 16, 32));
        float mnew = fmaxf(mrun, mloc);
        float corr = __expf(mrun - mnew);

        float p0[8], p1[8], ls = 0.f;
        #pragma unroll
        for (int i = 0; i < 8; ++i) {
            p0[i] = __expf(s0[i] - mnew);
            p1[i] = __expf(s1[i] - mnew);
            ls += p0[i] + p1[i];
        }
        ls += __shfl_xor(ls, 16, 32);
        lrun = lrun * corr + ls;
        mrun = mnew;
        #pragma unroll
        for (int d4 = 0; d4 < 4; ++d4) ot[d4] = ot[d4] * corr;

        // P^T B-fragment: lane needs keys 16*h2+j of its query; half comes from lane^16
        v16bf pb;
        #pragma unroll
        for (int i = 0; i < 8; ++i) {
            float w0 = __shfl_xor(p0[i], 16, 32);
            float w1 = __shfl_xor(p1[i], 16, 32);
            pb[i]     = tobf(h2 ? w1 : p0[i]);
            pb[i + 8] = tobf(h2 ? p1[i] : w0);
        }

        // O^T accumulate: 4 dim-slices of 16
        #pragma unroll
        for (int d4 = 0; d4 < 4; ++d4) {
            const __bf16* vrow = vbase + (size_t)(d4 * 16 + n) * NKP + kt;
            v8bf v0 = *(const v8bf*)(vrow + 16 * h2);
            v8bf v1 = *(const v8bf*)(vrow + 16 * h2 + 8);
            v16bf va;
            #pragma unroll
            for (int j = 0; j < 8; ++j) { va[j] = v0[j]; va[8 + j] = v1[j]; }
            ot[d4] = wmma_bf16(va, pb, ot[d4]);
        }
    }

    if (qok) {
        float inv = 1.f / lrun;
        size_t base = ((size_t)(bb * N_ + query)) * D_ + head * HD_;
        #pragma unroll
        for (int d4 = 0; d4 < 4; ++d4) {
            #pragma unroll
            for (int i = 0; i < 8; ++i)
                ob[base + d4 * 16 + 8 * h2 + i] = tobf(ot[d4][i] * inv);
        }
    }
}

// ---------------------------------------------------------------------------
// Kernel 3: output projection GEMM (bf16 WMMA, 16x32 tile/wave) + bias, fp32 out
// ---------------------------------------------------------------------------
__global__ __launch_bounds__(32)
void k_proj_gemm(const __bf16* __restrict__ ob, const __bf16* __restrict__ wb,
                 const float* __restrict__ bias, float* __restrict__ pf) {
    __shared__ __attribute__((aligned(16))) __bf16 Bs[32 * 32];
    const int lane = threadIdx.x;
    const int n  = lane & 15;
    const int h2 = lane >> 4;
    const int n0 = blockIdx.x * 32;     // [0,512)
    const int m0 = blockIdx.y * 16;     // [0,16400)
    v8f acc0 = {}, acc1 = {};

    for (int k0 = 0; k0 < D_; k0 += 32) {
        const __bf16* bp = wb + (size_t)(k0 + lane) * D_ + n0;
        v8bf r0 = *(const v8bf*)bp;
        v8bf r1 = *(const v8bf*)(bp + 8);
        v8bf r2 = *(const v8bf*)(bp + 16);
        v8bf r3 = *(const v8bf*)(bp + 24);
        __syncthreads();
        *(v8bf*)(&Bs[lane * 32])      = r0;
        *(v8bf*)(&Bs[lane * 32 + 8])  = r1;
        *(v8bf*)(&Bs[lane * 32 + 16]) = r2;
        *(v8bf*)(&Bs[lane * 32 + 24]) = r3;

        const __bf16* ap = ob + (size_t)(m0 + n) * D_ + k0;
        v8bf a0 = *(const v8bf*)(ap + 8 * h2);
        v8bf a1 = *(const v8bf*)(ap + 16 + 8 * h2);
        v16bf a;
        #pragma unroll
        for (int j = 0; j < 8; ++j) { a[j] = a0[j]; a[8 + j] = a1[j]; }

        __syncthreads();
        v16bf b0, b1;
        #pragma unroll
        for (int j = 0; j < 16; ++j) {
            b0[j] = Bs[((j & 8) * 2 + 8 * h2 + (j & 7)) * 32 + n];
            b1[j] = Bs[((j & 8) * 2 + 8 * h2 + (j & 7)) * 32 + 16 + n];
        }
        acc0 = wmma_bf16(a, b0, acc0);
        acc1 = wmma_bf16(a, b1, acc1);
    }

    float bv0 = bias[n0 + n];
    float bv1 = bias[n0 + 16 + n];
    #pragma unroll
    for (int i = 0; i < 8; ++i) {
        size_t row = (size_t)(m0 + 8 * h2 + i) * D_;
        pf[row + n0 + n]      = acc0[i] + bv0;
        pf[row + n0 + 16 + n] = acc1[i] + bv1;
    }
}

// ---------------------------------------------------------------------------
// Kernel 4: out = o + LayerNorm(o); one wave per row, shfl-xor reduction
// ---------------------------------------------------------------------------
__global__ __launch_bounds__(32)
void k_ln(const float* __restrict__ pf, const float* __restrict__ g,
          const float* __restrict__ be, float* __restrict__ out) {
    const int r = blockIdx.x;
    const int lane = threadIdx.x;
    const float4* row = (const float4*)(pf + (size_t)r * D_);
    float4 v[4];
    float s = 0.f, s2 = 0.f;
    #pragma unroll
    for (int j = 0; j < 4; ++j) {
        v[j] = row[j * 32 + lane];
        s  += v[j].x + v[j].y + v[j].z + v[j].w;
        s2 += v[j].x * v[j].x + v[j].y * v[j].y + v[j].z * v[j].z + v[j].w * v[j].w;
    }
    #pragma unroll
    for (int off = 16; off >= 1; off >>= 1) {
        s  += __shfl_xor(s,  off, 32);
        s2 += __shfl_xor(s2, off, 32);
    }
    const float mu   = s * (1.f / D_);
    const float var  = s2 * (1.f / D_) - mu * mu;
    const float rstd = rsqrtf(var + 1e-5f);
    const float4* g4 = (const float4*)g;
    const float4* b4 = (const float4*)be;
    float4* orow = (float4*)(out + (size_t)r * D_);
    #pragma unroll
    for (int j = 0; j < 4; ++j) {
        int c = j * 32 + lane;
        float4 gg = g4[c], bb = b4[c], o;
        o.x = v[j].x + (v[j].x - mu) * rstd * gg.x + bb.x;
        o.y = v[j].y + (v[j].y - mu) * rstd * gg.y + bb.y;
        o.z = v[j].z + (v[j].z - mu) * rstd * gg.z + bb.z;
        o.w = v[j].w + (v[j].w - mu) * rstd * gg.w + bb.w;
        orow[c] = o;
    }
}

// ---------------------------------------------------------------------------
extern "C" void kernel_launch(void* const* d_in, const int* in_sizes, int n_in,
                              void* d_out, int out_size, void* d_ws, size_t ws_size,
                              hipStream_t stream) {
    (void)in_sizes; (void)n_in; (void)out_size; (void)ws_size;
    const float* x     = (const float*)d_in[0];
    const float* wqkv  = (const float*)d_in[1];
    const float* wproj = (const float*)d_in[2];
    const float* bproj = (const float*)d_in[3];
    const float* gam   = (const float*)d_in[4];
    const float* bet   = (const float*)d_in[5];
    const float* mask  = (const float*)d_in[6];
    float* out = (float*)d_out;

    char* ws = (char*)d_ws;
    __bf16* wqkvb  = (__bf16*)(ws);                      //  1,572,864 B
    __bf16* wprojb = (__bf16*)(ws + 1572864);            //    524,288 B
    __bf16* qb     = (__bf16*)(ws + 2097152);            // 17,039,360 B
    __bf16* kb     = (__bf16*)(ws + 19136512);           // 17,301,504 B
    __bf16* vtb    = (__bf16*)(ws + 36438016);           // 17,301,504 B
    __bf16* ob     = (__bf16*)(ws + 53739520);           // 16,793,600 B
    float*  pf     = (float*)(ws + 70533120);            // 33,587,200 B  (total ~99.3 MB)

    k_convert<<<2048, 256, 0, stream>>>(wqkv, wproj, wqkvb, wprojb);

    dim3 g1(3 * D_ / 32, BN_ / 16);          // (48, 1025)
    k_qkv_gemm<<<g1, 32, 0, stream>>>(x, wqkvb, qb, kb, vtb);

    dim3 g2(NQP / 16, B_ * H_);              // (65, 128)
    k_attn<<<g2, 32, 0, stream>>>(qb, kb, vtb, mask, ob);

    dim3 g3(D_ / 32, BN_ / 16);              // (16, 1025)
    k_proj_gemm<<<g3, 32, 0, stream>>>(ob, wprojb, bproj, pf);

    k_ln<<<BN_, 32, 0, stream>>>(pf, gam, bet, out);
}
